// GCN_61607010893873
// MI455X (gfx1250) — matrix operand order repaired and checked by reading, not compiled, not run on hardware
//
#include <hip/hip_runtime.h>

// ---------------------------------------------------------------------------
// Vector / WMMA types
// ---------------------------------------------------------------------------
typedef __attribute__((ext_vector_type(16))) __bf16         bf16x16;
typedef __attribute__((ext_vector_type(16))) unsigned short ushort16;
typedef __attribute__((ext_vector_type(8)))  unsigned short ushort8;
typedef __attribute__((ext_vector_type(4)))  unsigned short ushort4v;
typedef __attribute__((ext_vector_type(4)))  float          floatx4;
typedef __attribute__((ext_vector_type(8)))  float          floatx8;

// round-to-nearest-even f32 -> bf16 bit pattern
__device__ __forceinline__ unsigned short f2bf(float f) {
    unsigned int u = __float_as_uint(f);
    unsigned int r = u + 0x7FFFu + ((u >> 16) & 1u);
    return (unsigned short)(r >> 16);
}

// ---------------------------------------------------------------------------
// Degree kernels: deg (stored in dinv buffer) = 1 (self loop) + in-degree
// ---------------------------------------------------------------------------
__global__ void deg_init_kernel(float* __restrict__ dinv, int n) {
    int i = blockIdx.x * blockDim.x + threadIdx.x;
    if (i < n) dinv[i] = 1.0f;
}

__global__ void deg_count_kernel(const int* __restrict__ dst, float* __restrict__ dinv, int nE) {
    int e = blockIdx.x * blockDim.x + threadIdx.x;
    if (e < nE) unsafeAtomicAdd(&dinv[dst[e]], 1.0f);
}

__global__ void deg_rsqrt_kernel(float* __restrict__ dinv, int n) {
    int i = blockIdx.x * blockDim.x + threadIdx.x;
    if (i < n) {
        float d = dinv[i];
        dinv[i] = d > 0.0f ? rsqrtf(d) : 0.0f;
    }
}

// ---------------------------------------------------------------------------
// Elementwise f32 -> bf16 (4 elements/thread; n multiple of 4)
// ---------------------------------------------------------------------------
__global__ void cvt_bf16_kernel(const float* __restrict__ in,
                                unsigned short* __restrict__ out, int n4) {
    int i = blockIdx.x * blockDim.x + threadIdx.x;
    if (i < n4) {
        floatx4 v = ((const floatx4*)in)[i];
        ushort4v o;
        o.x = f2bf(v.x); o.y = f2bf(v.y); o.z = f2bf(v.z); o.w = f2bf(v.w);
        ((ushort4v*)out)[i] = o;
    }
}

// W [K=128][HC] f32 row-major -> Wt [HC][128] bf16 (transposed, contiguous-K)
__global__ void cvt_w_kernel(const float* __restrict__ W,
                             unsigned short* __restrict__ Wt, int HC) {
    int i = blockIdx.x * blockDim.x + threadIdx.x;
    if (i < 128 * HC) {
        int k = i / HC;
        int c = i - k * HC;
        Wt[c * 128 + k] = f2bf(W[i]);
    }
}

// ---------------------------------------------------------------------------
// GEMM: hs[N x HC] = (X[N x 128] @ W[128 x HC]) * dinv[row]
// Inputs pre-converted to bf16: Xb [N][128], Wb [HC][128] (transposed).
// One wave per 16x16 output tile; HC/16 waves per block; 16 rows per block.
// Grid = N/16 (N = 50000 = 3125*16 exactly, no remainder).
// Per K-step: 2x16B loads for A, 2x16B loads for B, 1 WMMA.
// ---------------------------------------------------------------------------
template <int HC>
__global__ __launch_bounds__(HC * 2) void gemm_scaled_kernel(
    const unsigned short* __restrict__ Xb, const unsigned short* __restrict__ Wb,
    const float* __restrict__ dinv, float* __restrict__ hs) {

    const int tid  = threadIdx.x;
    const int wave = tid >> 5;
    const int lane = tid & 31;
    const int half = lane >> 4;       // 0: lanes 0-15, 1: lanes 16-31
    const int l16  = lane & 15;
    const int row0 = blockIdx.x * 16;
    const int colBase = wave * 16;

    // A: row (row0+l16); ISA layout: lane<16 holds K {kb..kb+7, kb+16..kb+23}
    // with kb = ks*32; lane>=16 same with kb = ks*32+8.
    const ushort8* pa = (const ushort8*)(Xb + (size_t)(row0 + l16) * 128);
    // B: col (colBase+l16); same K layout as A: lane<16 holds K {kb..kb+7,
    // kb+16..kb+23} with kb = ks*32; lane>=16 same with kb = ks*32+8.
    const ushort8* pb = (const ushort8*)(Wb + (size_t)(colBase + l16) * 128);

    floatx8 acc = {};
#pragma unroll
    for (int ks = 0; ks < 4; ++ks) {
        ushort8 alo = pa[ks * 4 + half];          // K = kb .. kb+7
        ushort8 ahi = pa[ks * 4 + half + 2];      // K = kb+16 .. kb+23
        ushort8 blo = pb[ks * 4 + half];          // K = kb .. kb+7
        ushort8 bhi = pb[ks * 4 + half + 2];      // K = kb+16 .. kb+23

        ushort16 au = __builtin_shufflevector(alo, ahi,
            0, 1, 2, 3, 4, 5, 6, 7, 8, 9, 10, 11, 12, 13, 14, 15);
        ushort16 bu = __builtin_shufflevector(blo, bhi,
            0, 1, 2, 3, 4, 5, 6, 7, 8, 9, 10, 11, 12, 13, 14, 15);

        acc = __builtin_amdgcn_wmma_f32_16x16x32_bf16(
            false, __builtin_bit_cast(bf16x16, au),
            false, __builtin_bit_cast(bf16x16, bu),
            (short)0, acc, false, false);
    }

    // C/D layout: VGPR v -> row v (lanes 0-15) / row v+8 (lanes 16-31),
    // col = lane&15. Fuse dinv[row] pre-scaling.
#pragma unroll
    for (int v = 0; v < 8; ++v) {
        int r = row0 + half * 8 + v;
        hs[(size_t)r * HC + colBase + l16] = acc[v] * dinv[r];
    }
}

// ---------------------------------------------------------------------------
// Edge aggregation: one wave per edge, acc[dst] += hs[src] (hs already scaled
// by dinv[src]). acc pre-initialized with hs (self-loop) via D2D memcpy.
// ---------------------------------------------------------------------------
template <int HC>
__global__ void agg_edges_kernel(const float* __restrict__ hs,
                                 const int* __restrict__ src,
                                 const int* __restrict__ dst,
                                 float* __restrict__ acc, int nE) {
    int gt = blockIdx.x * blockDim.x + threadIdx.x;
    int e = gt >> 5;
    if (e >= nE) return;
    int lane = gt & 31;
    int s = src[e];
    int d = dst[e];
    const float* ps = hs + (size_t)s * HC;
    float* pd = acc + (size_t)d * HC;
#pragma unroll
    for (int jj = 0; jj < HC / 32; ++jj) {
        int j = lane + jj * 32;
        unsafeAtomicAdd(&pd[j], ps[j]);
    }
}

// ---------------------------------------------------------------------------
// Finalize: io = prelu(dinv[node] * io + b[f])   (in place)
// ---------------------------------------------------------------------------
template <int HC>
__global__ void finalize_kernel(const float* __restrict__ dinv,
                                const float* __restrict__ b,
                                const float* __restrict__ aslope,
                                float* __restrict__ io, int n) {
    int i = blockIdx.x * blockDim.x + threadIdx.x;
    if (i >= n * HC) return;
    int node = i / HC;
    int f = i - node * HC;
    float v = io[i] * dinv[node] + b[f];
    float a = aslope[0];
    io[i] = v >= 0.0f ? v : a * v;
}

// ---------------------------------------------------------------------------
// Launch
// ---------------------------------------------------------------------------
extern "C" void kernel_launch(void* const* d_in, const int* in_sizes, int n_in,
                              void* d_out, int out_size, void* d_ws, size_t ws_size,
                              hipStream_t stream) {
    (void)n_in; (void)out_size; (void)ws_size;

    const float* x  = (const float*)d_in[0];    // [N,128]
    const int*   ei = (const int*)d_in[1];      // [2,E] flat
    const float* W1 = (const float*)d_in[2];    // [128,128]
    const float* b1 = (const float*)d_in[3];    // [128]
    const float* W2 = (const float*)d_in[4];    // [128,64]
    const float* b2 = (const float*)d_in[5];    // [64]
    const float* a  = (const float*)d_in[6];    // [1]

    const int N = in_sizes[0] / 128;
    const int E = in_sizes[1] / 2;
    const int* src = ei;
    const int* dst = ei + E;

    // workspace layout:
    //   floats : dinv[N] | hs1[N*128] | x2[N*128] | hs2[N*64]
    //   ushorts: xb[N*128] | x2b[N*128] | wb1[128*128] | wb2[64*128]
    float* dinv = (float*)d_ws;
    float* hs1  = dinv + N;
    float* x2   = hs1 + (size_t)N * 128;
    float* hs2  = x2  + (size_t)N * 128;
    unsigned short* xb  = (unsigned short*)(hs2 + (size_t)N * 64);
    unsigned short* x2b = xb  + (size_t)N * 128;
    unsigned short* wb1 = x2b + (size_t)N * 128;
    unsigned short* wb2 = wb1 + 128 * 128;
    float* out = (float*)d_out;                 // layer-2 accumulator lives in d_out

    const int TB = 256;

    // degrees -> dinv
    deg_init_kernel<<<(N + TB - 1) / TB, TB, 0, stream>>>(dinv, N);
    deg_count_kernel<<<(E + TB - 1) / TB, TB, 0, stream>>>(dst, dinv, E);
    deg_rsqrt_kernel<<<(N + TB - 1) / TB, TB, 0, stream>>>(dinv, N);

    // one-time bf16 conversions
    cvt_w_kernel<<<(128 * 128 + TB - 1) / TB, TB, 0, stream>>>(W1, wb1, 128);
    cvt_w_kernel<<<(128 * 64 + TB - 1) / TB, TB, 0, stream>>>(W2, wb2, 64);
    cvt_bf16_kernel<<<((size_t)N * 32 + TB - 1) / TB, TB, 0, stream>>>(x, xb, N * 32);

    // ---- layer 1 ----
    gemm_scaled_kernel<128><<<N / 16, 256, 0, stream>>>(xb, wb1, dinv, hs1);
    hipMemcpyAsync(x2, hs1, (size_t)N * 128 * sizeof(float),
                   hipMemcpyDeviceToDevice, stream);                 // self-loop init
    agg_edges_kernel<128><<<((size_t)E * 32 + TB - 1) / TB, TB, 0, stream>>>(
        hs1, src, dst, x2, E);
    finalize_kernel<128><<<((size_t)N * 128 + TB - 1) / TB, TB, 0, stream>>>(
        dinv, b1, a, x2, N);

    // ---- layer 2 ----
    cvt_bf16_kernel<<<((size_t)N * 32 + TB - 1) / TB, TB, 0, stream>>>(x2, x2b, N * 32);
    gemm_scaled_kernel<64><<<N / 16, 128, 0, stream>>>(x2b, wb2, dinv, hs2);
    hipMemcpyAsync(out, hs2, (size_t)N * 64 * sizeof(float),
                   hipMemcpyDeviceToDevice, stream);                 // self-loop init
    agg_edges_kernel<64><<<((size_t)E * 32 + TB - 1) / TB, TB, 0, stream>>>(
        hs2, src, dst, out, E);
    finalize_kernel<64><<<((size_t)N * 64 + TB - 1) / TB, TB, 0, stream>>>(
        dinv, b2, a, out, N);
}
